// QuantLlamaAttention_28226525069366
// MI455X (gfx1250) — 8-bit matrix operand order repaired and checked by reading, compiled, not run on hardware
//
#include <hip/hip_runtime.h>
#include <hip/hip_bf16.h>
#include <math.h>

#define HID    2048
#define NH     16
#define NKV    4
#define HD     128
#define GROUPS (NH / NKV)
#define BATCH  2
#define SEQ    2048
#define M_ROWS (BATCH * SEQ)

typedef __attribute__((ext_vector_type(8))) int v8i;

__device__ __forceinline__ float scale_from_bits(unsigned u) {
  return fmaxf(__uint_as_float(u) * (1.0f / 127.0f), 1e-8f);
}

__device__ __forceinline__ v8i load_a8(const signed char* p) {
  const int4 t0 = *(const int4*)(p + 0);
  const int4 t1 = *(const int4*)(p + 32);
  v8i r;
  r[0] = t0.x; r[1] = t0.y; r[2] = t0.z; r[3] = t0.w;
  r[4] = t1.x; r[5] = t1.y; r[6] = t1.z; r[7] = t1.w;
  return r;
}

__device__ __forceinline__ v8i load_b8(const signed char* p) {
  const int4 t0 = *(const int4*)(p + 0);
  const int4 t1 = *(const int4*)(p + 32);
  v8i r;
  r[0] = t0.x; r[1] = t0.y; r[2] = t0.z; r[3] = t0.w;
  r[4] = t1.x; r[5] = t1.y; r[6] = t1.z; r[7] = t1.w;
  return r;
}

__device__ __forceinline__ v8i wmma_iu8(v8i a, v8i b, v8i c) {
  return __builtin_amdgcn_wmma_i32_16x16x64_iu8(true, a, true, b, c, false, false);
}

__global__ void k_zero_scales(unsigned* s) {
  if (threadIdx.x < 16) s[threadIdx.x] = 0u;
}

__global__ void k_absmax(const float* __restrict__ x, size_t n, unsigned* slot) {
  __shared__ float red[256];
  float m = 0.0f;
  for (size_t i = (size_t)blockIdx.x * blockDim.x + threadIdx.x; i < n;
       i += (size_t)gridDim.x * blockDim.x)
    m = fmaxf(m, fabsf(x[i]));
  red[threadIdx.x] = m;
  __syncthreads();
  for (int s = 128; s > 0; s >>= 1) {
    if ((int)threadIdx.x < s)
      red[threadIdx.x] = fmaxf(red[threadIdx.x], red[threadIdx.x + s]);
    __syncthreads();
  }
  if (threadIdx.x == 0) atomicMax(slot, __float_as_uint(red[0]));
}

__global__ void k_quant(const float* __restrict__ x, signed char* __restrict__ q,
                        size_t n, const unsigned* __restrict__ slot) {
  const float inv = 1.0f / scale_from_bits(*slot);
  for (size_t i = (size_t)blockIdx.x * blockDim.x + threadIdx.x; i < n;
       i += (size_t)gridDim.x * blockDim.x) {
    int v = (int)rintf(x[i] * inv);
    v = v < -128 ? -128 : (v > 127 ? 127 : v);
    q[i] = (signed char)v;
  }
}

__global__ void k_quantT_v(const float* __restrict__ v, signed char* __restrict__ vt,
                           const unsigned* __restrict__ slot) {
  const float inv = 1.0f / scale_from_bits(*slot);
  const size_t n = (size_t)BATCH * NKV * SEQ * HD;
  for (size_t i = (size_t)blockIdx.x * blockDim.x + threadIdx.x; i < n;
       i += (size_t)gridDim.x * blockDim.x) {
    int d = (int)(i % HD);
    int s = (int)((i / HD) % SEQ);
    size_t bk = i / ((size_t)HD * SEQ);
    int q = (int)rintf(v[i] * inv);
    q = q < -128 ? -128 : (q > 127 ? 127 : q);
    vt[(bk * HD + d) * SEQ + s] = (signed char)q;
  }
}

__global__ void k_rope(float* __restrict__ t, int heads) {
  const size_t total = (size_t)BATCH * heads * SEQ * (HD / 2);
  size_t idx = (size_t)blockIdx.x * blockDim.x + threadIdx.x;
  if (idx >= total) return;
  int i = (int)(idx & 63);
  size_t tmp = idx >> 6;
  int s = (int)(tmp % SEQ);
  size_t bh = tmp / SEQ;
  float inv_freq = __expf(-((float)(2 * i) / (float)HD) * logf(10000.0f));
  float cs, sn;
  __sincosf((float)s * inv_freq, &sn, &cs);
  float* p = t + (bh * SEQ + (size_t)s) * HD;
  float a = p[i], b = p[i + 64];
  p[i]      = a * cs - b * sn;
  p[i + 64] = b * cs + a * sn;
}

__global__ void __launch_bounds__(32)
k_gemm_i8(const signed char* __restrict__ A, const signed char* __restrict__ Bt,
          float* __restrict__ out, int M, int N, int K,
          const unsigned* __restrict__ sc, int sa, int sb,
          int head_mode, int Sdim, int nheads, int hd) {
  const int lane = threadIdx.x;
  const int lo = lane & 15, hi = lane >> 4;
  const int tn = blockIdx.x, tm = blockIdx.y;

  const signed char* arow = A + (size_t)(tm * 16 + lo) * K + hi * 16;
  const signed char* brow = Bt + (size_t)(tn * 64 + lo) * K + hi * 16;

  v8i z = {0, 0, 0, 0, 0, 0, 0, 0};
  v8i acc[4];
#pragma unroll
  for (int i = 0; i < 4; ++i) acc[i] = z;

  v8i a_cur = load_a8(arow);
  v8i b_cur[4];
#pragma unroll
  for (int nt = 0; nt < 4; ++nt)
    b_cur[nt] = load_b8(brow + (size_t)(nt * 16) * K);

  for (int k0 = 64; k0 < K; k0 += 64) {
    __builtin_prefetch(arow + k0 + 192, 0, 3);
    v8i a_nxt = load_a8(arow + k0);
    v8i b_nxt[4];
#pragma unroll
    for (int nt = 0; nt < 4; ++nt)
      b_nxt[nt] = load_b8(brow + (size_t)(nt * 16) * K + k0);
#pragma unroll
    for (int nt = 0; nt < 4; ++nt)
      acc[nt] = wmma_iu8(a_cur, b_cur[nt], acc[nt]);
    a_cur = a_nxt;
#pragma unroll
    for (int nt = 0; nt < 4; ++nt) b_cur[nt] = b_nxt[nt];
  }
#pragma unroll
  for (int nt = 0; nt < 4; ++nt)
    acc[nt] = wmma_iu8(a_cur, b_cur[nt], acc[nt]);

  const float sf = scale_from_bits(sc[sa]) * scale_from_bits(sc[sb]);
#pragma unroll
  for (int nt = 0; nt < 4; ++nt) {
    const int n = tn * 64 + nt * 16 + lo;
#pragma unroll
    for (int r = 0; r < 8; ++r) {
      const int m = tm * 16 + r + 8 * hi;
      const float v = (float)acc[nt][r] * sf;
      if (head_mode) {
        int bb = m / Sdim, ss = m % Sdim;
        int hh = n / hd, dd = n % hd;
        out[(((size_t)bb * nheads + hh) * Sdim + ss) * hd + dd] = v;
      } else {
        out[(size_t)m * N + n] = v;
      }
    }
  }
}

__global__ void __launch_bounds__(32)
k_attn(const signed char* __restrict__ Qq, const signed char* __restrict__ Kq,
       const signed char* __restrict__ Vt, float* __restrict__ Of,
       const unsigned* __restrict__ sc) {
  __shared__ __align__(16) signed char sP[16 * 64];

  const int lane = threadIdx.x;
  const int lo = lane & 15, hi = lane >> 4;
  const int qt = blockIdx.x, h = blockIdx.y, b = blockIdx.z;
  const int kvh = h / GROUPS;

  const signed char* qrow =
      Qq + (((size_t)(b * NH + h)) * SEQ + qt * 16 + lo) * HD + hi * 16;
  const v8i aq0 = load_a8(qrow);
  const v8i aq1 = load_a8(qrow + 64);

  const float s_q = scale_from_bits(sc[5]);
  const float s_k = scale_from_bits(sc[6]);
  const float s_v = scale_from_bits(sc[7]);
  const float sscore = s_q * s_k * 0.08838834764831845f;

  const signed char* kbase = Kq + ((size_t)(b * NKV + kvh)) * SEQ * HD;
  const signed char* vbase = Vt + ((size_t)(b * NKV + kvh)) * HD * SEQ;

  v8i z = {0, 0, 0, 0, 0, 0, 0, 0};

  float m_run[8], l_run[8];
#pragma unroll
  for (int r = 0; r < 8; ++r) { m_run[r] = -1e30f; l_run[r] = 0.0f; }

  auto tile_update = [&](int kt, v8i b0, v8i b1) {
    v8i acc = z;
    acc = wmma_iu8(aq0, b0, acc);
    acc = wmma_iu8(aq1, b1, acc);
#pragma unroll
    for (int r = 0; r < 8; ++r) {
      const int qpos = qt * 16 + r + 8 * hi;
      const int kpos = kt * 16 + lo;
      float s = (kpos <= qpos) ? (float)acc[r] * sscore : -1e30f;
      float tmax = s;
      tmax = fmaxf(tmax, __shfl_xor(tmax, 1));
      tmax = fmaxf(tmax, __shfl_xor(tmax, 2));
      tmax = fmaxf(tmax, __shfl_xor(tmax, 4));
      tmax = fmaxf(tmax, __shfl_xor(tmax, 8));
      const float newm = fmaxf(m_run[r], tmax);
      float e = __expf(s - newm);
      e += __shfl_xor(e, 1);
      e += __shfl_xor(e, 2);
      e += __shfl_xor(e, 4);
      e += __shfl_xor(e, 8);
      l_run[r] = l_run[r] * __expf(m_run[r] - newm) + e;
      m_run[r] = newm;
    }
  };

  v8i bk0 = load_b8(kbase + (size_t)lo * HD + hi * 16);
  v8i bk1 = load_b8(kbase + (size_t)lo * HD + hi * 16 + 64);
  for (int kt = 0; kt < qt; ++kt) {
    const signed char* nk = kbase + (size_t)((kt + 1) * 16 + lo) * HD + hi * 16;
    v8i nb0 = load_b8(nk);
    v8i nb1 = load_b8(nk + 64);
    tile_update(kt, bk0, bk1);
    bk0 = nb0;
    bk1 = nb1;
  }
  tile_update(qt, bk0, bk1);

  float inv_l[8];
#pragma unroll
  for (int r = 0; r < 8; ++r) inv_l[r] = 1.0f / l_run[r];

  v8i oacc[8];
#pragma unroll
  for (int i = 0; i < 8; ++i) oacc[i] = z;

  const int nwin = qt / 4 + 1;
  for (int w = 0; w < nwin; ++w) {
    v8i vb[8];
#pragma unroll
    for (int dt = 0; dt < 8; ++dt)
      vb[dt] = load_b8(vbase + (size_t)(dt * 16 + lo) * SEQ + w * 64 + hi * 16);

#pragma unroll
    for (int sub = 0; sub < 4; ++sub) {
      const int kt = w * 4 + sub;
      if (kt <= qt) {
        const signed char* krow = kbase + (size_t)(kt * 16 + lo) * HD + hi * 16;
        v8i acc = z;
        acc = wmma_iu8(aq0, load_b8(krow), acc);
        acc = wmma_iu8(aq1, load_b8(krow + 64), acc);
#pragma unroll
        for (int r = 0; r < 8; ++r) {
          const int qpos = qt * 16 + r + 8 * hi;
          const int kpos = kt * 16 + lo;
          int pi = 0;
          if (kpos <= qpos) {
            float p = __expf((float)acc[r] * sscore - m_run[r]) * inv_l[r];
            pi = (int)rintf(p * 127.0f);
            pi = pi < 0 ? 0 : (pi > 127 ? 127 : pi);
          }
          sP[(r + 8 * hi) * 64 + sub * 16 + lo] = (signed char)pi;
        }
      } else {
#pragma unroll
        for (int r = 0; r < 8; ++r) sP[(r + 8 * hi) * 64 + sub * 16 + lo] = 0;
      }
    }
    __syncthreads();
    const v8i ap = load_a8(&sP[lo * 64 + hi * 16]);
    __syncthreads();
#pragma unroll
    for (int dt = 0; dt < 8; ++dt)
      oacc[dt] = wmma_iu8(ap, vb[dt], oacc[dt]);
  }

  const float so = s_v * (1.0f / 127.0f);
#pragma unroll
  for (int dt = 0; dt < 8; ++dt) {
#pragma unroll
    for (int r = 0; r < 8; ++r) {
      const int m = r + 8 * hi;
      const int d = dt * 16 + lo;
      const int qpos = qt * 16 + m;
      Of[(((size_t)b * SEQ + qpos) * NH + h) * HD + d] = (float)oacc[dt][r] * so;
    }
  }
}

extern "C" void kernel_launch(void* const* d_in, const int* in_sizes, int n_in,
                              void* d_out, int out_size, void* d_ws, size_t ws_size,
                              hipStream_t stream) {
  (void)in_sizes; (void)n_in; (void)out_size; (void)ws_size;
  const float* x  = (const float*)d_in[0];
  const float* wq = (const float*)d_in[3];
  const float* wk = (const float*)d_in[4];
  const float* wv = (const float*)d_in[5];
  const float* wo = (const float*)d_in[6];
  float* out = (float*)d_out;

  char* ws = (char*)d_ws;
  size_t off = 0;
  auto take = [&](size_t bytes) -> char* {
    char* p = ws + off;
    off = (off + bytes + 255) & ~(size_t)255;
    return p;
  };
  unsigned*    sc  = (unsigned*)take(64);
  signed char* Xq  = (signed char*)take((size_t)M_ROWS * HID);
  signed char* Wqq = (signed char*)take((size_t)NH * HD * HID);
  signed char* Wkq = (signed char*)take((size_t)NKV * HD * HID);
  signed char* Wvq = (signed char*)take((size_t)NKV * HD * HID);
  signed char* Woq = (signed char*)take((size_t)HID * NH * HD);
  float*       Qf  = (float*)take((size_t)M_ROWS * NH * HD * 4);
  float*       Kf  = (float*)take((size_t)M_ROWS * NKV * HD * 4);
  float*       Vf  = (float*)take((size_t)M_ROWS * NKV * HD * 4);
  signed char* Qq  = (signed char*)take((size_t)M_ROWS * NH * HD);
  signed char* Kq  = (signed char*)take((size_t)M_ROWS * NKV * HD);
  signed char* Vtq = (signed char*)take((size_t)M_ROWS * NKV * HD);
  float*       Off = (float*)take((size_t)M_ROWS * NH * HD * 4);
  signed char* Oq  = (signed char*)take((size_t)M_ROWS * NH * HD);

  const size_t nX  = (size_t)M_ROWS * HID;
  const size_t nWq = (size_t)NH * HD * HID;
  const size_t nWk = (size_t)NKV * HD * HID;
  const size_t nQ  = (size_t)M_ROWS * NH * HD;
  const size_t nK  = (size_t)M_ROWS * NKV * HD;

  k_zero_scales<<<1, 32, 0, stream>>>(sc);

  k_absmax<<<256, 256, 0, stream>>>(x,  nX,  sc + 0);
  k_absmax<<<128, 256, 0, stream>>>(wq, nWq, sc + 1);
  k_absmax<<<64,  256, 0, stream>>>(wk, nWk, sc + 2);
  k_absmax<<<64,  256, 0, stream>>>(wv, nWk, sc + 3);
  k_absmax<<<128, 256, 0, stream>>>(wo, nWq, sc + 4);

  k_quant<<<2048, 256, 0, stream>>>(x,  Xq,  nX,  sc + 0);
  k_quant<<<1024, 256, 0, stream>>>(wq, Wqq, nWq, sc + 1);
  k_quant<<<512,  256, 0, stream>>>(wk, Wkq, nWk, sc + 2);
  k_quant<<<512,  256, 0, stream>>>(wv, Wvq, nWk, sc + 3);
  k_quant<<<1024, 256, 0, stream>>>(wo, Woq, nWq, sc + 4);

  {
    dim3 gq(NH * HD / 64, M_ROWS / 16);
    k_gemm_i8<<<gq, 32, 0, stream>>>(Xq, Wqq, Qf, M_ROWS, NH * HD, HID,
                                     sc, 0, 1, 1, SEQ, NH, HD);
    dim3 gk(NKV * HD / 64, M_ROWS / 16);
    k_gemm_i8<<<gk, 32, 0, stream>>>(Xq, Wkq, Kf, M_ROWS, NKV * HD, HID,
                                     sc, 0, 2, 1, SEQ, NKV, HD);
    k_gemm_i8<<<gk, 32, 0, stream>>>(Xq, Wvq, Vf, M_ROWS, NKV * HD, HID,
                                     sc, 0, 3, 1, SEQ, NKV, HD);
  }

  k_rope<<<(unsigned)(((size_t)BATCH * NH * SEQ * 64 + 255) / 256), 256, 0, stream>>>(Qf, NH);
  k_rope<<<(unsigned)(((size_t)BATCH * NKV * SEQ * 64 + 255) / 256), 256, 0, stream>>>(Kf, NKV);

  k_absmax<<<256, 256, 0, stream>>>(Qf, nQ, sc + 5);
  k_absmax<<<128, 256, 0, stream>>>(Kf, nK, sc + 6);
  k_absmax<<<128, 256, 0, stream>>>(Vf, nK, sc + 7);
  k_quant<<<2048, 256, 0, stream>>>(Qf, Qq, nQ, sc + 5);
  k_quant<<<1024, 256, 0, stream>>>(Kf, Kq, nK, sc + 6);
  k_quantT_v<<<1024, 256, 0, stream>>>(Vf, Vtq, sc + 7);

  {
    dim3 ga(SEQ / 16, NH, BATCH);
    k_attn<<<ga, 32, 0, stream>>>(Qq, Kq, Vtq, Off, sc);
  }

  k_absmax<<<256, 256, 0, stream>>>(Off, nQ, sc + 8);
  k_quant<<<2048, 256, 0, stream>>>(Off, Oq, nQ, sc + 8);
  {
    dim3 go(HID / 64, M_ROWS / 16);
    k_gemm_i8<<<go, 32, 0, stream>>>(Oq, Woq, out, M_ROWS, HID, NH * HD,
                                     sc, 8, 4, 0, SEQ, NH, HD);
  }
}
